// SaltMHA_36155034698291
// MI455X (gfx1250) — matrix operand order repaired and checked by reading, not compiled, not run on hardware
//
#include <hip/hip_runtime.h>
#include <hip/hip_bf16.h>

// MI455X / gfx1250, wave32. WMMA f32_16x16x32_f16 for all GEMM-shaped work,
// VALU+LDS for the sequential RNN, W_o folded into the V projection,
// double-buffered global_load_async_to_lds staging of K/V tiles in attention.

#define BB 8
#define TT 1024
#define EE 1024
#define HH 16
#define DD 64

// Padded LDS tile strides (in halfs): keep 16B alignment, spread banks.
#define KSTRIDE 72   // 32 rows x 144B  (K tile: keys x d)
#define VSTRIDE 40   // 64 rows x 80B   (V tile: d x keys)

typedef __attribute__((ext_vector_type(16))) _Float16 v16h;
typedef __attribute__((ext_vector_type(8)))  _Float16 v8h;
typedef __attribute__((ext_vector_type(8)))  float    v8f;

__device__ __forceinline__ v8f wmma_f16(v16h a, v16h b, v8f c) {
  // (neg_a, A, neg_b, B, c_mod, C, reuse_a, reuse_b)
  return __builtin_amdgcn_wmma_f32_16x16x32_f16(false, a, false, b, (short)0, c,
                                                false, false);
}

// A-matrix (16x32 f16) fragment from an f32 row (convert on the fly).
// Per-lane K runs: lower lanes dblk+0..15, upper lanes dblk+16..31.
__device__ __forceinline__ v16h load_a_frag_f32(const float* __restrict__ rowbase,
                                                int dblk, bool lo) {
  const int ko = lo ? 0 : 16;
  const float* p0 = rowbase + dblk + ko;
  const float* p1 = rowbase + dblk + 8 + ko;
  v16h a;
#pragma unroll
  for (int i = 0; i < 8; ++i) {
    a[i]     = (_Float16)p0[i];
    a[i + 8] = (_Float16)p1[i];
  }
  return a;
}

// 16B async copy global -> LDS (raw LDS byte offset in low 32 bits of a
// generic __shared__ pointer, per ISA flat->LDS aperture mapping).
__device__ __forceinline__ void async_cp16(uint32_t lds_byte, const void* g) {
  asm volatile("global_load_async_to_lds_b128 %0, %1, off"
               :: "v"(lds_byte), "v"(g)
               : "memory");
}

// ---------------------------------------------------------------------------
// Kernel 1: weight prep.  W_c = W_ih + W_hh, b_c = b_ih + b_hh,
// WqT[h][e][d] = W_q[h][d][e] (f16), WvoT[h][e][d] = (W_v[h] @ W_o[h])^T (f16).
// ---------------------------------------------------------------------------
__global__ void prep_weights(const float* __restrict__ Wq,
                             const float* __restrict__ Wv,
                             const float* __restrict__ Wo,
                             const float* __restrict__ Wih,
                             const float* __restrict__ Whh,
                             const float* __restrict__ bih,
                             const float* __restrict__ bhh,
                             _Float16* __restrict__ WqT,
                             _Float16* __restrict__ WvoT,
                             float* __restrict__ Wc,
                             float* __restrict__ bc) {
  __shared__ float sWv[DD * DD];
  __shared__ float sWo[DD * DD];
  const int h = blockIdx.x;
  const int tid = threadIdx.x;  // 256
  for (int i = tid; i < DD * DD; i += 256) {
    sWv[i] = Wv[h * DD * DD + i];
    sWo[i] = Wo[h * DD * DD + i];
    Wc[h * DD * DD + i] = Wih[h * DD * DD + i] + Whh[h * DD * DD + i];
    // i = d*64 + e  ->  WqT[e][d]
    WqT[h * DD * DD + (i % DD) * DD + (i / DD)] = (_Float16)Wq[h * DD * DD + i];
  }
  if (tid < DD) bc[h * DD + tid] = bih[h * DD + tid] + bhh[h * DD + tid];
  __syncthreads();
  for (int i = tid; i < DD * DD; i += 256) {
    const int d = i / DD, e = i % DD;
    float acc = 0.f;
#pragma unroll 8
    for (int f = 0; f < DD; ++f) acc += sWv[d * DD + f] * sWo[f * DD + e];
    WvoT[h * DD * DD + e * DD + d] = (_Float16)acc;  // transpose on store
  }
}

// ---------------------------------------------------------------------------
// Kernel 2: sequential RNN.  One block per head (512 thr: b = tid/64, e = tid%64).
// h1 = tanh(W_ih k0 + b_c);  h_{t+1} = tanh(W_c h_t + b_c).  nw stored f16.
// ---------------------------------------------------------------------------
__global__ void rnn_kernel(const float* __restrict__ target,
                           const float* __restrict__ Wih,
                           const float* __restrict__ Wc,
                           const float* __restrict__ bc,
                           _Float16* __restrict__ nw) {
  __shared__ float sWih[DD * DD];
  __shared__ float sWc[DD * DD];
  __shared__ float sbc[DD];
  __shared__ float hbuf[BB][DD];
  const int h = blockIdx.x;
  const int tid = threadIdx.x;  // 512
  const int b = tid >> 6, e = tid & 63;
  for (int i = tid; i < DD * DD; i += 512) {
    sWih[i] = Wih[h * DD * DD + i];  // [e][d] per einsum 'hed'
    sWc[i]  = Wc[h * DD * DD + i];
  }
  if (tid < DD) sbc[tid] = bc[h * DD + tid];
  hbuf[b][e] = target[(b * TT + 0) * EE + h * DD + e];  // k0
  __syncthreads();

  float acc = sbc[e];
#pragma unroll 8
  for (int d = 0; d < DD; ++d) acc += sWih[e * DD + d] * hbuf[b][d];
  float hv = tanhf(acc);
  __syncthreads();
  hbuf[b][e] = hv;
  nw[((b * HH + h) * TT + 0) * DD + e] = (_Float16)hv;
  __syncthreads();

  for (int t = 1; t < TT; ++t) {
    float a2 = sbc[e];
#pragma unroll 8
    for (int d = 0; d < DD; ++d) a2 += sWc[e * DD + d] * hbuf[b][d];
    const float hn = tanhf(a2);
    __syncthreads();
    hbuf[b][e] = hn;
    nw[((b * HH + h) * TT + t) * DD + e] = (_Float16)hn;
    __syncthreads();
  }
}

// ---------------------------------------------------------------------------
// Kernel 3: projections via WMMA, two passes (Q then V) to avoid VGPR spills.
// q = src_h @ W_q  -> qbuf[bh][t][d] (f16);  v = tgt_h @ W_vo -> vT[bh][d][t].
// ---------------------------------------------------------------------------
__global__ void proj_kernel(const float* __restrict__ src,
                            const float* __restrict__ tgt,
                            const _Float16* __restrict__ WqT,
                            const _Float16* __restrict__ WvoT,
                            _Float16* __restrict__ qbuf,
                            _Float16* __restrict__ vT) {
  const int lane = threadIdx.x & 31;
  const int wave = threadIdx.x >> 5;
  const int bh = blockIdx.x >> 3;
  const int b = bh >> 4, h = bh & 15;
  const int t0 = (blockIdx.x & 7) * 128 + wave * 16;
  const int m = lane & 15;
  const bool lo = lane < 16;
  const int trow = t0 + m;
  const int rbase = lo ? 0 : 8;

  // ---- pass 1: Q ----
  {
    v16h bw[4][2];
#pragma unroll
    for (int et = 0; et < 4; ++et)
#pragma unroll
      for (int db = 0; db < 2; ++db)
        bw[et][db] = *(const v16h*)(WqT + (h * DD + et * 16 + m) * DD + db * 32 + (lo ? 0 : 16));
    v8f c[4];
#pragma unroll
    for (int et = 0; et < 4; ++et)
#pragma unroll
      for (int r = 0; r < 8; ++r) c[et][r] = 0.f;
#pragma unroll
    for (int db = 0; db < 2; ++db) {
      const v16h a = load_a_frag_f32(src + (size_t)(b * TT + trow) * EE + h * DD, db * 32, lo);
#pragma unroll
      for (int et = 0; et < 4; ++et) c[et] = wmma_f16(a, bw[et][db], c[et]);
    }
#pragma unroll
    for (int et = 0; et < 4; ++et)
#pragma unroll
      for (int r = 0; r < 8; ++r)
        qbuf[(size_t)(bh * TT + t0 + rbase + r) * DD + et * 16 + m] = (_Float16)c[et][r];
  }

  // ---- pass 2: V (with W_o folded), stored transposed [d][t] ----
  {
    v16h bw[4][2];
#pragma unroll
    for (int et = 0; et < 4; ++et)
#pragma unroll
      for (int db = 0; db < 2; ++db)
        bw[et][db] = *(const v16h*)(WvoT + (h * DD + et * 16 + m) * DD + db * 32 + (lo ? 0 : 16));
    v8f c[4];
#pragma unroll
    for (int et = 0; et < 4; ++et)
#pragma unroll
      for (int r = 0; r < 8; ++r) c[et][r] = 0.f;
#pragma unroll
    for (int db = 0; db < 2; ++db) {
      const v16h a = load_a_frag_f32(tgt + (size_t)(b * TT + trow) * EE + h * DD, db * 32, lo);
#pragma unroll
      for (int et = 0; et < 4; ++et) c[et] = wmma_f16(a, bw[et][db], c[et]);
    }
#pragma unroll
    for (int et = 0; et < 4; ++et)
#pragma unroll
      for (int r = 0; r < 8; ++r)
        vT[(size_t)(bh * DD + et * 16 + m) * TT + t0 + rbase + r] = (_Float16)c[et][r];
  }
}

// ---------------------------------------------------------------------------
// Kernel 4: flash attention, transposed formulation, LDS-staged K/V tiles.
// S^T = nw @ Q^T  (queries on lanes -> per-lane softmax stats),
// O^T = V^T @ P^T (C->B layout fixup via shfl_xor(16)).  Output is final
// (W_o folded into V).  logit scale = 1/(sqrt(D)*sqrt(T)) = 1/256.
// K/V tiles (4KB each per 32 keys) are shared by all 8 waves of the block and
// streamed with double-buffered global_load_async_to_lds_b128.
// ---------------------------------------------------------------------------
__global__ void attn_kernel(const _Float16* __restrict__ qbuf,
                            const _Float16* __restrict__ nw,
                            const _Float16* __restrict__ vT,
                            float* __restrict__ out) {
  __shared__ __align__(16) _Float16 sK[2][32 * KSTRIDE];
  __shared__ __align__(16) _Float16 sV[2][64 * VSTRIDE];

  const int tid = threadIdx.x;  // 256
  const int lane = tid & 31;
  const int wave = tid >> 5;
  const int bh = blockIdx.x >> 3;
  const int b = bh >> 4, h = bh & 15;
  const int q0 = (blockIdx.x & 7) * 128 + wave * 16;
  const int m = lane & 15;
  const bool lo = lane < 16;
  const int ko = lo ? 0 : 16;

  const uint32_t sK_base = (uint32_t)(uintptr_t)&sK[0][0];
  const uint32_t sV_base = (uint32_t)(uintptr_t)&sV[0][0];

  // Staging: K tile 32 rows x 128B (8 chunks), V tile 64 rows x 64B (4 chunks).
  const int kr = tid >> 3, kc = tid & 7;   // one 16B chunk per thread
  const int vr = tid >> 2, vc = tid & 3;

#define STAGE(bufi, kkv)                                                        \
  do {                                                                          \
    async_cp16(sK_base + (uint32_t)((bufi) * 32 * KSTRIDE * 2 + kr * KSTRIDE * 2 + kc * 16), \
               nw + (size_t)(bh * TT + (kkv) + kr) * DD + kc * 8);              \
    async_cp16(sV_base + (uint32_t)((bufi) * 64 * VSTRIDE * 2 + vr * VSTRIDE * 2 + vc * 16), \
               vT + (size_t)(bh * DD + vr) * TT + (kkv) + vc * 8);              \
  } while (0)

  // Q^T as B fragments: lane col = query m; K = d block.
  v16h bq[2];
#pragma unroll
  for (int db = 0; db < 2; ++db)
    bq[db] = *(const v16h*)(qbuf + (size_t)(bh * TT + q0 + m) * DD + db * 32 + (lo ? 0 : 16));

  float mstat = -1e30f, lstat = 0.f;
  v8f acc[4];
#pragma unroll
  for (int dt = 0; dt < 4; ++dt)
#pragma unroll
    for (int r = 0; r < 8; ++r) acc[dt][r] = 0.f;

  STAGE(0, 0);  // prologue: ASYNCcnt = 2 per wave

  for (int kk = 0; kk < TT; kk += 32) {
    const int cur = (kk >> 5) & 1;
    const int nxt = cur ^ 1;
    // Issue next tile (wraps harmlessly on the last iteration), then wait for
    // the current tile: completions are in-order, 2 ops/stage per wave.
    STAGE(nxt, (kk + 32) & (TT - 1));
    asm volatile("s_wait_asynccnt 0x2" ::: "memory");
    __syncthreads();

    // ---- S^T = K_tile @ Q^T ----
    v8f s0, s1;
#pragma unroll
    for (int r = 0; r < 8; ++r) { s0[r] = 0.f; s1[r] = 0.f; }
#pragma unroll
    for (int db = 0; db < 2; ++db) {
      const _Float16* k0p = &sK[cur][(m) * KSTRIDE + db * 32 + ko];
      const _Float16* k1p = &sK[cur][(16 + m) * KSTRIDE + db * 32 + ko];
      v16h a0, a1;
      {
        v8h x0 = *(const v8h*)k0p, x1 = *(const v8h*)(k0p + 8);
        v8h y0 = *(const v8h*)k1p, y1 = *(const v8h*)(k1p + 8);
#pragma unroll
        for (int i = 0; i < 8; ++i) {
          a0[i] = x0[i]; a0[i + 8] = x1[i];
          a1[i] = y0[i]; a1[i + 8] = y1[i];
        }
      }
      s0 = wmma_f16(a0, bq[db], s0);
      s1 = wmma_f16(a1, bq[db], s1);
    }

    // ---- online softmax over key axis (per-lane query stats) ----
    const float sc = 1.0f / 256.0f;
    float p0[8], p1[8];
    float cmax = -1e30f;
#pragma unroll
    for (int r = 0; r < 8; ++r) {
      p0[r] = s0[r] * sc;
      p1[r] = s1[r] * sc;
      cmax = fmaxf(cmax, fmaxf(p0[r], p1[r]));
    }
    cmax = fmaxf(cmax, __shfl_xor(cmax, 16, 32));  // combine key halves
    const float mnew = fmaxf(mstat, cmax);
    const float f = __expf(mstat - mnew);
    mstat = mnew;
    lstat *= f;
#pragma unroll
    for (int dt = 0; dt < 4; ++dt)
#pragma unroll
      for (int r = 0; r < 8; ++r) acc[dt][r] *= f;

    float lsum = 0.f;
#pragma unroll
    for (int r = 0; r < 8; ++r) {
      p0[r] = __expf(p0[r] - mnew);
      p1[r] = __expf(p1[r] - mnew);
      lsum += p0[r] + p1[r];
    }
    lsum += __shfl_xor(lsum, 16, 32);
    lstat += lsum;

    // C-layout S^T -> B-layout P^T: swap row-halves across lane halves.
    v16h bp;
#pragma unroll
    for (int j = 0; j < 8; ++j) {
      const float x0 = __shfl_xor(p0[j], 16, 32);
      const float x1 = __shfl_xor(p1[j], 16, 32);
      bp[j]     = (_Float16)(lo ? p0[j] : x1);
      bp[j + 8] = (_Float16)(lo ? x0 : p1[j]);
    }

    // ---- O^T += V^T @ P^T ----
#pragma unroll
    for (int dt = 0; dt < 4; ++dt) {
      const _Float16* vrow = &sV[cur][(dt * 16 + m) * VSTRIDE];
      const v8h v0 = *(const v8h*)(vrow + ko);
      const v8h v1 = *(const v8h*)(vrow + 8 + ko);
      v16h av;
#pragma unroll
      for (int i = 0; i < 8; ++i) { av[i] = v0[i]; av[i + 8] = v1[i]; }
      acc[dt] = wmma_f16(av, bp, acc[dt]);
    }

    __syncthreads();  // protect buffers before next stage overwrites
  }
#undef STAGE

  const float linv = 1.0f / lstat;
  const int rbase = lo ? 0 : 8;
  const int t = q0 + m;  // query column on this lane
#pragma unroll
  for (int dt = 0; dt < 4; ++dt)
#pragma unroll
    for (int r = 0; r < 8; ++r) {
      const int d = dt * 16 + rbase + r;
      out[(size_t)(b * TT + t) * EE + h * DD + d] = acc[dt][r] * linv;
    }
}

// ---------------------------------------------------------------------------
extern "C" void kernel_launch(void* const* d_in, const int* in_sizes, int n_in,
                              void* d_out, int out_size, void* d_ws, size_t ws_size,
                              hipStream_t stream) {
  (void)in_sizes; (void)n_in; (void)out_size; (void)ws_size;
  const float* src = (const float*)d_in[0];
  const float* tgt = (const float*)d_in[1];
  const float* Wq  = (const float*)d_in[2];
  const float* Wv  = (const float*)d_in[3];
  const float* Wo  = (const float*)d_in[4];
  const float* Wih = (const float*)d_in[5];
  const float* Whh = (const float*)d_in[6];
  const float* bih = (const float*)d_in[7];
  const float* bhh = (const float*)d_in[8];

  char* ws = (char*)d_ws;
  size_t off = 0;
  _Float16* qbuf = (_Float16*)(ws + off); off += (size_t)BB * HH * TT * DD * 2;
  _Float16* nw   = (_Float16*)(ws + off); off += (size_t)BB * HH * TT * DD * 2;
  _Float16* vT   = (_Float16*)(ws + off); off += (size_t)BB * HH * TT * DD * 2;
  _Float16* WqT  = (_Float16*)(ws + off); off += (size_t)HH * DD * DD * 2;
  _Float16* WvoT = (_Float16*)(ws + off); off += (size_t)HH * DD * DD * 2;
  float* Wc = (float*)(ws + off); off += (size_t)HH * DD * DD * 4;
  float* bc = (float*)(ws + off); off += (size_t)HH * DD * 4;

  prep_weights<<<HH, 256, 0, stream>>>(Wq, Wv, Wo, Wih, Whh, bih, bhh,
                                       WqT, WvoT, Wc, bc);
  rnn_kernel<<<HH, 512, 0, stream>>>(tgt, Wih, Wc, bc, nw);
  proj_kernel<<<BB * HH * (TT / 128), 256, 0, stream>>>(src, tgt, WqT, WvoT,
                                                        qbuf, vT);
  attn_kernel<<<BB * HH * (TT / 128), 256, 0, stream>>>(qbuf, nw, vT,
                                                        (float*)d_out);
}
